// TransformerModel_55009941127919
// MI455X (gfx1250) — matrix operand order repaired and checked by reading, not compiled, not run on hardware
//
#include <hip/hip_runtime.h>
#include <hip/hip_bf16.h>

// ---------------------------------------------------------------------------
// ViT forward for MI455X (gfx1250, wave32). All GEMMs via v_wmma_f32_16x16x32_f16.
// B=64, S=197, D=128, H=8 (DH=16), FF=3072, L=8, 14x14 patches of 16x16x3.
// ---------------------------------------------------------------------------

typedef __attribute__((ext_vector_type(16))) _Float16 v16h;
typedef __attribute__((ext_vector_type(8)))  float    v8f;

constexpr int GBM  = 64;  // block tile M
constexpr int GBN  = 64;  // block tile N (2 wave-col groups x 32)
constexpr int GBK  = 32;  // K step
constexpr int LDSR = 40;  // LDS row stride in halves (80B: 16B-aligned, staggers banks)

union Half8 { uint4 u; _Float16 h[8]; };
union Frag  { uint4 u[2]; v16h v; };

// Generic tiled WMMA GEMM:  C = alpha * (A @ B) [+ bias] [+ C_old] [relu]
//   A(m,k) at Abase[m*lda + k]        (k contiguous)
//   B(k,n) at Bbase[k*bsk + n*bsn]    (either operand order)
//   C(m,n) at Cbase[m*ldc + n]
// Batched over blockIdx.z with z = zb*Hdiv + zh; per-operand (zb,zh) strides.
// flags: bit0 = accumulate into existing C (residual), bit1 = relu after bias.
__global__ __launch_bounds__(256)
void wmma_gemm(const float* __restrict__ A,  long lda, long zAb, long zAh,
               const float* __restrict__ Bs, long bsk, long bsn, long zBb, long zBh,
               const float* __restrict__ bias,
               float*       __restrict__ C,  long ldc, long zCb, long zCh,
               int M, int N, int K, int Hdiv, float alpha, int flags)
{
    __shared__ _Float16 sA[GBM * LDSR];
    __shared__ _Float16 sB[GBN * LDSR];

    const int z  = blockIdx.z;
    const int zb = z / Hdiv, zh = z % Hdiv;
    const float* Ab = A  + (long)zb * zAb + (long)zh * zAh;
    const float* Bb = Bs + (long)zb * zBb + (long)zh * zBh;
    float*       Cb = C  + (long)zb * zCb + (long)zh * zCh;

    const int m0  = blockIdx.y * GBM;
    const int n0  = blockIdx.x * GBN;
    const int tid = threadIdx.x;
    const int lane = tid & 31;
    const int wid  = tid >> 5;
    const int wm = wid >> 1;      // 0..3 : 16-row sub-tile
    const int wn = wid & 1;       // 0..1 : 32-col group
    const int r  = lane & 15;
    const int hi = lane >> 4;

    // cooperative staging: each thread handles 8 contiguous k for one row
    const int arow = tid >> 2;            // 0..63 (m)
    const int ak0  = (tid & 3) * 8;       // 0,8,16,24
    const int brow = tid >> 2;            // 0..63 (n)
    const int bk0  = (tid & 3) * 8;

    // uniform (scalar-branch) path selectors — no per-lane exec churn
    const bool aVec  = ((lda & 3) == 0) && ((((unsigned long)(const void*)Ab) & 15) == 0);
    const bool bVec  = (bsk == 1) && ((bsn & 3) == 0) &&
                       ((((unsigned long)(const void*)Bb) & 15) == 0);
    const bool fullM = (m0 + GBM <= M);
    const bool fullN = (n0 + GBN <= N);

    v8f acc0 = {};
    v8f acc1 = {};

    for (int kk = 0; kk < K; kk += GBK) {
        const bool fullK = (kk + GBK <= K);

        // ---------------- stage A tile (GBM x GBK) -> f16 LDS ----------------
        {
            const int   mg  = m0 + arow;
            const float* src = Ab + (long)mg * lda + kk + ak0;
            Half8 pk;
            if (fullM && fullK) {
                if (kk + GBK < K) __builtin_prefetch(src + GBK, 0, 1);
                if (aVec) {
                    const float4 a0 = ((const float4*)src)[0];
                    const float4 a1 = ((const float4*)src)[1];
                    pk.h[0] = (_Float16)a0.x; pk.h[1] = (_Float16)a0.y;
                    pk.h[2] = (_Float16)a0.z; pk.h[3] = (_Float16)a0.w;
                    pk.h[4] = (_Float16)a1.x; pk.h[5] = (_Float16)a1.y;
                    pk.h[6] = (_Float16)a1.z; pk.h[7] = (_Float16)a1.w;
                } else {
                    #pragma unroll
                    for (int i = 0; i < 8; ++i) pk.h[i] = (_Float16)src[i];
                }
            } else {
                // branchless edge: clamp address to a valid location, cndmask value
                #pragma unroll
                for (int i = 0; i < 8; ++i) {
                    const bool ok = (mg < M) && (kk + ak0 + i < K);
                    const float x = *(ok ? (src + i) : Ab);
                    pk.h[i] = ok ? (_Float16)x : (_Float16)0.0f;
                }
            }
            *(uint4*)&sA[arow * LDSR + ak0] = pk.u;
        }
        // ---------------- stage B tile as [n][k] (GBN x GBK) ----------------
        {
            const int ng = n0 + brow;
            Half8 pk;
            if (fullN && fullK) {
                if (bVec) {
                    const float* src = Bb + (long)ng * bsn + kk + bk0;
                    if (kk + GBK < K) __builtin_prefetch(src + GBK, 0, 1);
                    const float4 b0 = ((const float4*)src)[0];
                    const float4 b1 = ((const float4*)src)[1];
                    pk.h[0] = (_Float16)b0.x; pk.h[1] = (_Float16)b0.y;
                    pk.h[2] = (_Float16)b0.z; pk.h[3] = (_Float16)b0.w;
                    pk.h[4] = (_Float16)b1.x; pk.h[5] = (_Float16)b1.y;
                    pk.h[6] = (_Float16)b1.z; pk.h[7] = (_Float16)b1.w;
                } else {
                    #pragma unroll
                    for (int i = 0; i < 8; ++i)
                        pk.h[i] = (_Float16)Bb[(long)(kk + bk0 + i) * bsk + (long)ng * bsn];
                }
            } else {
                #pragma unroll
                for (int i = 0; i < 8; ++i) {
                    const int  kg = kk + bk0 + i;
                    const bool ok = (ng < N) && (kg < K);
                    const float* p = ok ? (Bb + (long)kg * bsk + (long)ng * bsn) : Bb;
                    const float x = *p;
                    pk.h[i] = ok ? (_Float16)x : (_Float16)0.0f;
                }
            }
            *(uint4*)&sB[brow * LDSR + bk0] = pk.u;
        }
        __syncthreads();

        // ---------------- fragments in contiguous k order, same for A and B ----------------
        // A 16x32: lane(m=r,hi): e0..15 -> K=hi*16+e
        Frag fa, fb0, fb1;
        const _Float16* pa = &sA[(wm * 16 + r) * LDSR];
        fa.u[0] = *(const uint4*)(pa + hi * 16);
        fa.u[1] = *(const uint4*)(pa + hi * 16 + 8);
        // B 32x16: lane(n=r,hi): e0..15 -> K=hi*16+e (contiguous in sB[n][k])
        const _Float16* pb0 = &sB[(wn * 32 + r) * LDSR + hi * 16];
        const _Float16* pb1 = pb0 + 16 * LDSR;
        fb0.u[0] = *(const uint4*)(pb0);
        fb0.u[1] = *(const uint4*)(pb0 + 8);
        fb1.u[0] = *(const uint4*)(pb1);
        fb1.u[1] = *(const uint4*)(pb1 + 8);

        acc0 = __builtin_amdgcn_wmma_f32_16x16x32_f16(
            false, fa.v, false, fb0.v, (short)0, acc0, false, false);
        acc1 = __builtin_amdgcn_wmma_f32_16x16x32_f16(
            false, fa.v, false, fb1.v, (short)0, acc1, false, false);
        __syncthreads();
    }

    // ---------------- store: VGPR j -> M = hi*8 + j ----------------
    const int ng0 = n0 + wn * 32 + r;
    const int ng1 = ng0 + 16;
    #pragma unroll
    for (int j = 0; j < 8; ++j) {
        const int mg = m0 + wm * 16 + hi * 8 + j;
        if (mg < M) {
            if (ng0 < N) {
                float val = acc0[j] * alpha;
                if (bias)      val += bias[ng0];
                if (flags & 1) val += Cb[(long)mg * ldc + ng0];
                if (flags & 2) val  = fmaxf(val, 0.0f);
                Cb[(long)mg * ldc + ng0] = val;
            }
            if (ng1 < N) {
                float val = acc1[j] * alpha;
                if (bias)      val += bias[ng1];
                if (flags & 1) val += Cb[(long)mg * ldc + ng1];
                if (flags & 2) val  = fmaxf(val, 0.0f);
                Cb[(long)mg * ldc + ng1] = val;
            }
        }
    }
}

// ---------------- row softmax, one wave32 per row (n <= 224) ----------------
__global__ __launch_bounds__(256)
void softmax_rows(float* __restrict__ P, int rows, int n)
{
    const int row = blockIdx.x * 8 + (threadIdx.x >> 5);
    if (row >= rows) return;
    const int lane = threadIdx.x & 31;
    float* p = P + (long)row * n;
    float buf[7];
    float mx = -3.4e38f;
    int c = 0;
    for (int i = lane; i < n; i += 32) { float x = p[i]; buf[c++] = x; mx = fmaxf(mx, x); }
    #pragma unroll
    for (int o = 16; o > 0; o >>= 1) mx = fmaxf(mx, __shfl_xor(mx, o));
    float s = 0.0f; c = 0;
    for (int i = lane; i < n; i += 32) { float e = __expf(buf[c] - mx); buf[c++] = e; s += e; }
    #pragma unroll
    for (int o = 16; o > 0; o >>= 1) s += __shfl_xor(s, o);
    const float inv = 1.0f / s;
    c = 0;
    for (int i = lane; i < n; i += 32) p[i] = buf[c++] * inv;
}

// ---------------- layernorm over D=128, one wave32 per row ----------------
__global__ __launch_bounds__(256)
void layernorm_rows(const float* __restrict__ X, float* __restrict__ Y,
                    const float* __restrict__ g, const float* __restrict__ b, int rows)
{
    const int row = blockIdx.x * 8 + (threadIdx.x >> 5);
    if (row >= rows) return;
    const int lane = threadIdx.x & 31;
    const float4 x = ((const float4*)(X + (long)row * 128))[lane];
    float s = x.x + x.y + x.z + x.w;
    #pragma unroll
    for (int o = 16; o > 0; o >>= 1) s += __shfl_xor(s, o);
    const float mean = s * (1.0f / 128.0f);
    const float d0 = x.x - mean, d1 = x.y - mean, d2 = x.z - mean, d3 = x.w - mean;
    float vs = d0 * d0 + d1 * d1 + d2 * d2 + d3 * d3;
    #pragma unroll
    for (int o = 16; o > 0; o >>= 1) vs += __shfl_xor(vs, o);
    const float rstd = rsqrtf(vs * (1.0f / 128.0f) + 1e-12f);
    const float4 gv = ((const float4*)g)[lane];
    const float4 bv = ((const float4*)b)[lane];
    float4 y;
    y.x = d0 * rstd * gv.x + bv.x;
    y.y = d1 * rstd * gv.y + bv.y;
    y.z = d2 * rstd * gv.z + bv.z;
    y.w = d3 * rstd * gv.w + bv.w;
    ((float4*)(Y + (long)row * 128))[lane] = y;
}

// ---------------- im2col for 16x16 patches of (B,3,224,224) ----------------
__global__ void im2col_patches(const float* __restrict__ x, float* __restrict__ out, int total)
{
    const int t = blockIdx.x * 256 + threadIdx.x;
    if (t >= total) return;
    const int k = t % 768;  const int rest = t / 768;
    const int p = rest % 196; const int b = rest / 196;
    const int c = k >> 8; const int rem = k & 255;
    const int i = rem >> 4, j = rem & 15;
    const int py = p / 14, px = p % 14;
    out[t] = x[(((long)b * 3 + c) * 224 + py * 16 + i) * 224 + px * 16 + j];
}

// ---------------- h = concat(cls, z) + pos ----------------
__global__ void assemble_h(const float* __restrict__ z, const float* __restrict__ cls,
                           const float* __restrict__ pos, float* __restrict__ h, int total)
{
    const int t = blockIdx.x * 256 + threadIdx.x;
    if (t >= total) return;
    const int d = t & 127; const int rest = t >> 7;
    const int s = rest % 197; const int b = rest / 197;
    const float v = (s == 0) ? cls[d] : z[((long)b * 196 + (s - 1)) * 128 + d];
    h[t] = v + pos[s * 128 + d];
}

// ---------------- final heads: 5 dots of length 128, wave per batch ----------------
__global__ void heads_kernel(const float* __restrict__ ln,
                             const float* __restrict__ w_hf,  const float* __restrict__ b_hf,
                             const float* __restrict__ w_htc, const float* __restrict__ b_htc,
                             const float* __restrict__ w_reg, const float* __restrict__ b_reg,
                             float* __restrict__ out)
{
    const int b = blockIdx.x;
    const int lane = threadIdx.x;
    const float* cls = ln + (long)b * 197 * 128;   // row s=0
    float hf = 0, ht = 0, r0 = 0, r1 = 0, r2 = 0;
    for (int d = lane; d < 128; d += 32) {
        const float c = cls[d];
        hf += c * w_hf[d];  ht += c * w_htc[d];
        r0 += c * w_reg[d]; r1 += c * w_reg[128 + d]; r2 += c * w_reg[256 + d];
    }
    #pragma unroll
    for (int o = 16; o > 0; o >>= 1) {
        hf += __shfl_xor(hf, o); ht += __shfl_xor(ht, o);
        r0 += __shfl_xor(r0, o); r1 += __shfl_xor(r1, o); r2 += __shfl_xor(r2, o);
    }
    if (lane == 0) {
        out[b]               = hf + b_hf[0];
        out[64 + b]          = ht + b_htc[0];
        out[128 + b * 3 + 0] = r0 + b_reg[0];
        out[128 + b * 3 + 1] = r1 + b_reg[1];
        out[128 + b * 3 + 2] = r2 + b_reg[2];
    }
}

extern "C" void kernel_launch(void* const* d_in, const int* in_sizes, int n_in,
                              void* d_out, int out_size, void* d_ws, size_t ws_size,
                              hipStream_t stream)
{
    (void)in_sizes; (void)n_in; (void)out_size; (void)ws_size;
    // setup_inputs() dict order, depth-first:
    const float* x       = (const float*)d_in[0];
    const float* patch_w = (const float*)d_in[1];
    const float* patch_b = (const float*)d_in[2];
    const float* cls     = (const float*)d_in[3];
    const float* pos     = (const float*)d_in[4];
    const float* ln1_g   = (const float*)d_in[5];
    const float* ln1_b   = (const float*)d_in[6];
    const float* wq      = (const float*)d_in[7];
    const float* bq      = (const float*)d_in[8];
    const float* wk      = (const float*)d_in[9];
    const float* bk      = (const float*)d_in[10];
    const float* wv      = (const float*)d_in[11];
    const float* bv      = (const float*)d_in[12];
    const float* wo      = (const float*)d_in[13];
    const float* bo      = (const float*)d_in[14];
    const float* ln2_g   = (const float*)d_in[15];
    const float* ln2_b   = (const float*)d_in[16];
    const float* w1      = (const float*)d_in[17];
    const float* b1      = (const float*)d_in[18];
    const float* w2      = (const float*)d_in[19];
    const float* b2      = (const float*)d_in[20];
    const float* lnf_g   = (const float*)d_in[21];
    const float* lnf_b   = (const float*)d_in[22];
    const float* w_hf    = (const float*)d_in[23];
    const float* b_hf    = (const float*)d_in[24];
    const float* w_htc   = (const float*)d_in[25];
    const float* b_htc   = (const float*)d_in[26];
    const float* w_reg   = (const float*)d_in[27];
    const float* b_reg   = (const float*)d_in[28];
    float* out = (float*)d_out;

    constexpr long TOK  = 64L * 197;      // 12608 rows
    constexpr long TOKD = TOK * 128;
    float* h   = (float*)d_ws;
    float* lnb = h   + TOKD;
    float* q   = lnb + TOKD;
    float* k   = q   + TOKD;
    float* v   = k   + TOKD;
    float* ctx = v   + TOKD;
    float* ff  = ctx + TOKD;              // 12608*3072 f32; also hosts im2col (12544*768)

    // --- stem ---
    const int tot_ic = 64 * 196 * 768;
    im2col_patches<<<(tot_ic + 255) / 256, 256, 0, stream>>>(x, ff, tot_ic);
    {   // z(q) = im2col @ patch_w^T + patch_b    M=12544 N=128 K=768
        dim3 g(128 / GBN, (64 * 196) / GBM, 1);
        wmma_gemm<<<g, 256, 0, stream>>>(ff, 768, 0, 0,  patch_w, 1, 768, 0, 0,
                                         patch_b,  q, 128, 0, 0,
                                         64 * 196, 128, 768, 1, 1.0f, 0);
    }
    const int tot_h = (int)TOKD;
    assemble_h<<<(tot_h + 255) / 256, 256, 0, stream>>>(q, cls, pos, h, tot_h);

    const long SS   = 197L * 197;
    const long BHSS = 64L * 8 * SS;
    float* attn_base = out + 320;               // after hf(64)+htc(64)+regime(192)

    const dim3 gTok(128 / GBN, (int)(TOK / GBM), 1);   // N=128, M=12608
    for (int l = 0; l < 8; ++l) {
        const float* g1  = ln1_g + l * 128;  const float* be1 = ln1_b + l * 128;
        const float* g2  = ln2_g + l * 128;  const float* be2 = ln2_b + l * 128;
        const float* wql = wq + (long)l * 128 * 128;  const float* bql = bq + l * 128;
        const float* wkl = wk + (long)l * 128 * 128;  const float* bkl = bk + l * 128;
        const float* wvl = wv + (long)l * 128 * 128;  const float* bvl = bv + l * 128;
        const float* wol = wo + (long)l * 128 * 128;  const float* bol = bo + l * 128;
        const float* w1l = w1 + (long)l * 3072 * 128; const float* b1l = b1 + l * 3072;
        const float* w2l = w2 + (long)l * 128 * 3072; const float* b2l = b2 + l * 128;
        float* attn_l = attn_base + (long)l * BHSS;

        layernorm_rows<<<(int)(TOK / 8), 256, 0, stream>>>(h, lnb, g1, be1, (int)TOK);

        wmma_gemm<<<gTok, 256, 0, stream>>>(lnb, 128, 0, 0,  wql, 1, 128, 0, 0, bql,
                                            q, 128, 0, 0, (int)TOK, 128, 128, 1, 1.0f, 0);
        wmma_gemm<<<gTok, 256, 0, stream>>>(lnb, 128, 0, 0,  wkl, 1, 128, 0, 0, bkl,
                                            k, 128, 0, 0, (int)TOK, 128, 128, 1, 1.0f, 0);
        wmma_gemm<<<gTok, 256, 0, stream>>>(lnb, 128, 0, 0,  wvl, 1, 128, 0, 0, bvl,
                                            v, 128, 0, 0, (int)TOK, 128, 128, 1, 1.0f, 0);

        // scores[b,h] = q_bh(197x16) @ k_bh^T / 4 ; batched z = b*8+h (512)
        {
            dim3 g((197 + GBN - 1) / GBN, (197 + GBM - 1) / GBM, 512);
            wmma_gemm<<<g, 256, 0, stream>>>(q, 128, 197L * 128, 16,
                                             k, 1, 128, 197L * 128, 16,
                                             nullptr,
                                             attn_l, 197, 8 * SS, SS,
                                             197, 197, 16, 8, 0.25f, 0);
        }
        softmax_rows<<<(int)(512L * 197 / 8), 256, 0, stream>>>(attn_l, 512 * 197, 197);

        // ctx[b,h] = probs(197x197) @ v_bh(197x16)
        {
            dim3 g(1, (197 + GBM - 1) / GBM, 512);
            wmma_gemm<<<g, 256, 0, stream>>>(attn_l, 197, 8 * SS, SS,
                                             v, 128, 1, 197L * 128, 16,
                                             nullptr,
                                             ctx, 128, 197L * 128, 16,
                                             197, 16, 197, 8, 1.0f, 0);
        }
        // h += ctx @ wo^T + bo
        wmma_gemm<<<gTok, 256, 0, stream>>>(ctx, 128, 0, 0,  wol, 1, 128, 0, 0, bol,
                                            h, 128, 0, 0, (int)TOK, 128, 128, 1, 1.0f, 1);

        layernorm_rows<<<(int)(TOK / 8), 256, 0, stream>>>(h, lnb, g2, be2, (int)TOK);

        // ff = relu(lnb @ w1^T + b1)   M=12608 N=3072 K=128
        {
            dim3 g(3072 / GBN, (int)(TOK / GBM), 1);
            wmma_gemm<<<g, 256, 0, stream>>>(lnb, 128, 0, 0,  w1l, 1, 128, 0, 0, b1l,
                                             ff, 3072, 0, 0, (int)TOK, 3072, 128, 1, 1.0f, 2);
        }
        // h += ff @ w2^T + b2          M=12608 N=128 K=3072
        wmma_gemm<<<gTok, 256, 0, stream>>>(ff, 3072, 0, 0,  w2l, 1, 3072, 0, 0, b2l,
                                            h, 128, 0, 0, (int)TOK, 128, 3072, 1, 1.0f, 1);
    }

    layernorm_rows<<<(int)(TOK / 8), 256, 0, stream>>>(h, lnb, lnf_g, lnf_b, (int)TOK);
    heads_kernel<<<64, 32, 0, stream>>>(lnb, w_hf, b_hf, w_htc, b_htc, w_reg, b_reg, out);
}
